// MPSLanguageModel_89816356094451
// MI455X (gfx1250) — matrix operand order repaired and checked by reading, not compiled, not run on hardware
//
#include <hip/hip_runtime.h>
#include <hip/hip_bf16.h>

// ---------------------------------------------------------------------------
// MPS language model loss for MI455X (gfx1250, wave32, WMMA).
//
// logits[r=(t*8+b), v] = sum_{i,j} (h_mod[r,i]*mr[t,j]) * core[i,v,j] + bias[v]
//   == A(4096x4096, bf16) @ C2(4096x32000) with C2[(i*64+j), v] = core[i][v][j]
// GEMM runs on v_wmma_f32_16x16x32_bf16; softmax stats fused in the epilogue.
// A-tiles stream in via the Tensor Data Mover (tensor_load_to_lds, 6-arg form).
// ---------------------------------------------------------------------------

#define VV     32000
#define CHI    64
#define TT     512
#define BB     8
#define ROWS   4096            // T*B
#define KK     4096            // CHI*CHI
#define MB     128             // rows per workgroup
#define NB     128             // vocab cols per workgroup
#define KB     32              // k per WMMA step
#define NBLK   (VV / NB)       // 250
#define LDSROW 17              // 16 data dwords + 1 pad dword per 32-elem bf16 row

typedef __attribute__((ext_vector_type(16))) __bf16 v16bf;
typedef __attribute__((ext_vector_type(8)))  float  v8f;
typedef __attribute__((ext_vector_type(4)))  unsigned v4u_t;
typedef __attribute__((ext_vector_type(8)))  int      v8i_t;
typedef __attribute__((ext_vector_type(4)))  int      v4i_t;

#if defined(__has_builtin)
#if __has_builtin(__builtin_amdgcn_tensor_load_to_lds) && __has_builtin(__builtin_amdgcn_s_wait_tensorcnt)
#define USE_TDM 1
#endif
#endif
#ifndef USE_TDM
#define USE_TDM 0
#endif

static __device__ __forceinline__ unsigned pack_bf16x2(float lo, float hi) {
  unsigned ul = __float_as_uint(lo);
  unsigned uh = __float_as_uint(hi);
  ul = (ul + 0x7FFFu + ((ul >> 16) & 1u)) >> 16;   // RNE f32 -> bf16
  uh = (uh + 0x7FFFu + ((uh >> 16) & 1u)) >> 16;
  return (uh << 16) | (ul & 0xFFFFu);
}

// ---------------- Kernel 1: position-MLP modulators ml/mr ------------------
__global__ void k_modulators(const float* __restrict__ pos_embed,
                             const float* __restrict__ W1, const float* __restrict__ b1,
                             const float* __restrict__ W2, const float* __restrict__ b2,
                             float* __restrict__ ml, float* __restrict__ mr) {
  int t = blockIdx.x, c = threadIdx.x;              // 128 threads
  __shared__ float pos[64];
  __shared__ float hm[128];
  if (c < 64) pos[c] = pos_embed[t * 64 + c];
  __syncthreads();
  float a = b1[c];
  for (int p = 0; p < 64; ++p) a += pos[p] * W1[p * 128 + c];
  hm[c] = 0.5f * a * (1.0f + erff(a * 0.70710678118654752f));   // exact GELU
  __syncthreads();
  float m = b2[c];
  for (int q = 0; q < 128; ++q) m += hm[q] * W2[q * 128 + c];
  float val = 1.0f + 0.5f * tanhf(m);
  if (c < 64) ml[t * 64 + c] = val;
  else        mr[t * 64 + (c - 64)] = val;
}

// ---------------- Kernel 2: sequential recurrence per batch lane -----------
__global__ void k_recurrence(const int* __restrict__ input_ids,
                             const float* __restrict__ core,
                             const float* __restrict__ h0,
                             const float* __restrict__ ml, const float* __restrict__ mr,
                             const float* __restrict__ ln_g, const float* __restrict__ ln_b,
                             float* __restrict__ hmod) {
  int b = blockIdx.x, tid = threadIdx.x;            // 256 threads
  int j = tid & 63, i4 = tid >> 6;
  __shared__ float h[64], hmv[64], part[4][64], rsum[64], rsq[64];
  if (tid < 64) h[tid] = h0[tid];
  __syncthreads();
  for (int t = 0; t < TT; ++t) {
    if (tid < 64) {
      float v = h[tid] * ml[t * 64 + tid];
      hmv[tid] = v;
      hmod[(size_t)(t * BB + b) * 64 + tid] = v;
    }
    __syncthreads();
    int x = input_ids[b * TT + t];
    const float* cp = core + (size_t)x * CHI + (size_t)j;
    float p = 0.f;
    for (int i = i4; i < 64; i += 4) p += hmv[i] * cp[(size_t)i * ((size_t)VV * CHI)];
    part[i4][j] = p;
    __syncthreads();
    float hnew = 0.f;
    if (tid < 64) {
      hnew = (part[0][j] + part[1][j] + part[2][j] + part[3][j]) * mr[t * 64 + j];
      rsum[j] = hnew; rsq[j] = hnew * hnew;
    }
    __syncthreads();
    for (int s = 32; s > 0; s >>= 1) {
      if (tid < s) { rsum[tid] += rsum[tid + s]; rsq[tid] += rsq[tid + s]; }
      __syncthreads();
    }
    if (tid < 64) {
      float mu  = rsum[0] * (1.0f / 64.0f);
      float var = rsq[0] * (1.0f / 64.0f) - mu * mu;
      h[tid] = (hnew - mu) * rsqrtf(var + 1e-5f) * ln_g[tid] + ln_b[tid];
    }
    __syncthreads();
  }
}

// ---------------- Kernel 3: build A (bf16 4096x4096) -----------------------
__global__ void k_buildA(const float* __restrict__ hmod, const float* __restrict__ mr,
                         unsigned* __restrict__ A_u) {
  const int n = ROWS * (KK / 2);
  for (int u = blockIdx.x * blockDim.x + threadIdx.x; u < n; u += gridDim.x * blockDim.x) {
    int r = u >> 11, k = (u & 2047) << 1;
    int i = k >> 6, jj = k & 63, t = r >> 3;
    float hv = hmod[r * 64 + i];
    A_u[u] = pack_bf16x2(hv * mr[t * 64 + jj], hv * mr[t * 64 + jj + 1]);
  }
}

// ---------------- WMMA fragment loaders (ISA §7.12.2 layouts) --------------
static __device__ __forceinline__ v16bf load_fragA(const unsigned* shA, int mt, int lane) {
  int m = lane & 15, half = lane >> 4;
  const unsigned* row = shA + (mt * 16 + m) * LDSROW;
  union { v16bf v; unsigned u[8]; } f;
#pragma unroll
  for (int v = 0; v < 4; ++v) f.u[v] = row[v + 4 * half];          // K = 2v+8*half
#pragma unroll
  for (int v = 4; v < 8; ++v) f.u[v] = row[8 + (v - 4) + 4 * half]; // K = 16+2(v-4)+8*half
  return f.v;
}
static __device__ __forceinline__ v16bf load_fragB(const unsigned* shB, int nt, int lane) {
  int n = lane & 15, half = lane >> 4;
  const unsigned* row = shB + (nt * 16 + n) * LDSROW;              // LDS holds B as [n][k]
  union { v16bf v; unsigned u[8]; } f;
#pragma unroll
  for (int p = 0; p < 8; ++p) f.u[p] = row[(p & 3) + 4 * half + 8 * (p >> 2)]; // K = 2(p&3)+8*half+16*(p>>2)
  return f.v;
}

// ---------------- Kernel 4: WMMA GEMM + fused softmax stats ----------------
__global__ __launch_bounds__(256) void k_gemm(
    const unsigned* __restrict__ A_u, const float* __restrict__ core,
    const float* __restrict__ bias, const int* __restrict__ target_ids,
    float* __restrict__ stats_m, float* __restrict__ stats_s,
    float* __restrict__ tgt_logit) {
  __shared__ unsigned shA[MB * LDSROW];
  __shared__ unsigned shB[NB * LDSROW];
  __shared__ float ls_rm[MB][2], ls_rs[MB][2], ls_rmf[MB];
  __shared__ int ls_tgt[MB];

  const int tid = threadIdx.x, l = tid & 31, w = tid >> 5;
  const int wm = w & 3, wn = w >> 2;                // 4 waves along M, 2 along N
  const int nb = blockIdx.x, mb = blockIdx.y;
  const int v0 = nb * NB, r0 = mb * MB;
  const int half = l >> 4, lc = l & 15;

  v8f acc[2][4];
  const v8f vz = {0.f, 0.f, 0.f, 0.f, 0.f, 0.f, 0.f, 0.f};
#pragma unroll
  for (int mi = 0; mi < 2; ++mi)
#pragma unroll
    for (int ni = 0; ni < 4; ++ni) acc[mi][ni] = vz;

#if USE_TDM
  const unsigned lds_off = (unsigned)(size_t)(&shA[0]);
#endif

  for (int kb = 0; kb < KK / KB; ++kb) {
    const int k0 = kb * KB;
#if USE_TDM
    if (w == 0) {  // one TDM issue per block: A tile 128 rows x 32 bf16, padded rows
      unsigned long long ga =
          (unsigned long long)(size_t)A_u + ((size_t)r0 * KK + k0) * 2ull;
      v4u_t g0 = {1u,                                   // count=1
                  lds_off,                              // lds_addr
                  (unsigned)(ga & 0xffffffffu),
                  (unsigned)((ga >> 32) & 0x1ffffffu) | (2u << 30)};  // type=2
      v8i_t g1;
      g1[0] = (int)((1u << 16) | (1u << 20) | (3u << 22)); // 2B elems, pad 1dw/16dw
      g1[1] = (int)((unsigned)(KK & 0xffff) << 16);        // tensor_dim0 lo16
      g1[2] = (int)((unsigned)(ROWS & 0xffff) << 16);      // dim0 hi=0 | tensor_dim1 lo16
      g1[3] = (int)(32u << 16);                            // dim1 hi=0 | tile_dim0=32
      g1[4] = (int)128;                                    // tile_dim1=128, tile_dim2=0
      g1[5] = (int)KK;                                     // tensor_dim0_stride lo32
      g1[6] = 0; g1[7] = 0;
      v4i_t gz = {0, 0, 0, 0};                             // groups 2/3 unused (2D)
      v8i_t g4z = {0, 0, 0, 0, 0, 0, 0, 0};                // trailing group (clang-23 form)
      __builtin_amdgcn_tensor_load_to_lds(g0, g1, gz, gz, g4z, 0);
    }
#else
#pragma unroll
    for (int it = 0; it < 8; ++it) {                 // fallback: manual A tile load
      int f = tid + it * 256, m = f >> 4, c = f & 15;
      shA[m * LDSROW + c] = A_u[(size_t)(r0 + m) * (KK / 2) + (k0 >> 1) + c];
    }
#endif
    // B tile: core f32 (i0, v0..v0+127, j0..j0+31) -> bf16 in LDS as [n][k]
    const int i0 = k0 >> 6, j0 = k0 & 63;
    const float* cbase = core + (size_t)i0 * ((size_t)VV * CHI) + (size_t)v0 * CHI + j0;
#pragma unroll
    for (int it = 0; it < 4; ++it) {
      int f = tid + it * 256, v = f >> 3, q = f & 7;
      const float4 d = *reinterpret_cast<const float4*>(cbase + (size_t)v * CHI + q * 4);
      shB[v * LDSROW + q * 2]     = pack_bf16x2(d.x, d.y);
      shB[v * LDSROW + q * 2 + 1] = pack_bf16x2(d.z, d.w);
    }
#if USE_TDM
    if (w == 0) __builtin_amdgcn_s_wait_tensorcnt(0);
#endif
    __syncthreads();

    v16bf fa[2], fb[4];
#pragma unroll
    for (int mi = 0; mi < 2; ++mi) fa[mi] = load_fragA(shA, wm * 2 + mi, l);
#pragma unroll
    for (int ni = 0; ni < 4; ++ni) fb[ni] = load_fragB(shB, wn * 4 + ni, l);
#pragma unroll
    for (int mi = 0; mi < 2; ++mi)
#pragma unroll
      for (int ni = 0; ni < 4; ++ni)
        acc[mi][ni] = __builtin_amdgcn_wmma_f32_16x16x32_bf16(
            false, fa[mi], false, fb[ni], (short)0, acc[mi][ni], false, false);
    __syncthreads();
  }

  // ---- fused softmax epilogue: bias, target scatter, row max / sum-exp ----
  if (tid < MB) {
    int rr = r0 + tid;
    ls_tgt[tid] = target_ids[(rr & 7) * TT + (rr >> 3)];  // r = t*8+b -> tgt[b*T+t]
  }
  float bcol[4];
#pragma unroll
  for (int ni = 0; ni < 4; ++ni) bcol[ni] = bias[v0 + (wn * 4 + ni) * 16 + lc];
#pragma unroll
  for (int mi = 0; mi < 2; ++mi)
#pragma unroll
    for (int ni = 0; ni < 4; ++ni)
#pragma unroll
      for (int a = 0; a < 8; ++a) acc[mi][ni][a] += bcol[ni];
  __syncthreads();

  float rm[2][8];
#pragma unroll
  for (int mi = 0; mi < 2; ++mi) {
    int rbase = (wm * 2 + mi) * 16 + half * 8;
#pragma unroll
    for (int a = 0; a < 8; ++a) {
      int rl = rbase + a, tg = ls_tgt[rl];
      float mx = -3.4e38f;
#pragma unroll
      for (int ni = 0; ni < 4; ++ni) {
        float x = acc[mi][ni][a];
        mx = fmaxf(mx, x);
        if (tg == v0 + (wn * 4 + ni) * 16 + lc) tgt_logit[r0 + rl] = x;
      }
#pragma unroll
      for (int msk = 1; msk < 16; msk <<= 1) mx = fmaxf(mx, __shfl_xor(mx, msk, 16));
      rm[mi][a] = mx;
    }
  }
  if (lc == 0)
#pragma unroll
    for (int mi = 0; mi < 2; ++mi)
#pragma unroll
      for (int a = 0; a < 8; ++a) ls_rm[(wm * 2 + mi) * 16 + half * 8 + a][wn] = rm[mi][a];
  __syncthreads();
  if (tid < MB) ls_rmf[tid] = fmaxf(ls_rm[tid][0], ls_rm[tid][1]);
  __syncthreads();
#pragma unroll
  for (int mi = 0; mi < 2; ++mi) {
    int rbase = (wm * 2 + mi) * 16 + half * 8;
#pragma unroll
    for (int a = 0; a < 8; ++a) {
      int rl = rbase + a;
      float m = ls_rmf[rl], s = 0.f;
#pragma unroll
      for (int ni = 0; ni < 4; ++ni) s += __expf(acc[mi][ni][a] - m);
#pragma unroll
      for (int msk = 1; msk < 16; msk <<= 1) s += __shfl_xor(s, msk, 16);
      if (lc == 0) ls_rs[rl][wn] = s;
    }
  }
  __syncthreads();
  if (tid < MB) {
    stats_m[(size_t)nb * ROWS + r0 + tid] = ls_rmf[tid];
    stats_s[(size_t)nb * ROWS + r0 + tid] = ls_rs[tid][0] + ls_rs[tid][1];
  }
}

// ---------------- Kernel 5: merge per-block stats -> per-row loss ----------
__global__ void k_rowloss(const float* __restrict__ stats_m, const float* __restrict__ stats_s,
                          const float* __restrict__ tgt_logit, float* __restrict__ rowloss) {
  int r = blockIdx.x * blockDim.x + threadIdx.x;
  if (r >= ROWS) return;
  float m = -3.4e38f;
  for (int nb = 0; nb < NBLK; ++nb) m = fmaxf(m, stats_m[nb * ROWS + r]);
  float s = 0.f;
  for (int nb = 0; nb < NBLK; ++nb) s += stats_s[nb * ROWS + r] * expf(stats_m[nb * ROWS + r] - m);
  rowloss[r] = m + logf(s) - tgt_logit[r];
}

// ---------------- Kernel 6: mean over rows -> scalar loss ------------------
__global__ void k_final(const float* __restrict__ rowloss, float* __restrict__ out) {
  __shared__ float red[256];
  float s = 0.f;
  for (int r = threadIdx.x; r < ROWS; r += 256) s += rowloss[r];
  red[threadIdx.x] = s;
  __syncthreads();
  for (int st = 128; st > 0; st >>= 1) {
    if ((int)threadIdx.x < st) red[threadIdx.x] += red[threadIdx.x + st];
    __syncthreads();
  }
  if (threadIdx.x == 0) out[0] = red[0] * (1.0f / (float)ROWS);
}

// ---------------- Workspace layout (all sizes multiple of 256B) ------------
static const size_t WS_ML   = 0;                         // 512*64 f32
static const size_t WS_MR   = WS_ML   + 131072;
static const size_t WS_HMOD = WS_MR   + 131072;          // 4096*64 f32
static const size_t WS_TGT  = WS_HMOD + 1048576;         // 4096 f32
static const size_t WS_RL   = WS_TGT  + 16384;           // 4096 f32
static const size_t WS_SM   = WS_RL   + 16384;           // 250*4096 f32
static const size_t WS_SS   = WS_SM   + 4096000;
static const size_t WS_A    = WS_SS   + 4096000;         // 4096*4096 bf16 = 32MB
// total ~= 43 MB

extern "C" void kernel_launch(void* const* d_in, const int* in_sizes, int n_in,
                              void* d_out, int out_size, void* d_ws, size_t ws_size,
                              hipStream_t stream) {
  const int*   input_ids   = (const int*)d_in[0];
  const int*   target_ids  = (const int*)d_in[1];
  const float* core        = (const float*)d_in[2];
  const float* h0          = (const float*)d_in[3];
  const float* pos_embed   = (const float*)d_in[4];
  const float* W1          = (const float*)d_in[5];
  const float* b1          = (const float*)d_in[6];
  const float* W2          = (const float*)d_in[7];
  const float* b2          = (const float*)d_in[8];
  const float* output_bias = (const float*)d_in[9];
  const float* ln_g        = (const float*)d_in[10];
  const float* ln_b        = (const float*)d_in[11];

  char* ws = (char*)d_ws;
  float*    ml      = (float*)(ws + WS_ML);
  float*    mr      = (float*)(ws + WS_MR);
  float*    hmod    = (float*)(ws + WS_HMOD);
  float*    tgtl    = (float*)(ws + WS_TGT);
  float*    rowloss = (float*)(ws + WS_RL);
  float*    sm      = (float*)(ws + WS_SM);
  float*    ss      = (float*)(ws + WS_SS);
  unsigned* A_u     = (unsigned*)(ws + WS_A);
  float*    out     = (float*)d_out;

  k_modulators<<<TT, 128, 0, stream>>>(pos_embed, W1, b1, W2, b2, ml, mr);
  k_recurrence<<<BB, 256, 0, stream>>>(input_ids, core, h0, ml, mr, ln_g, ln_b, hmod);
  k_buildA<<<4096, 256, 0, stream>>>(hmod, mr, A_u);
  k_gemm<<<dim3(NBLK, ROWS / MB), 256, 0, stream>>>(A_u, core, output_bias, target_ids,
                                                    sm, ss, tgtl);
  k_rowloss<<<ROWS / 256, 256, 0, stream>>>(sm, ss, tgtl, rowloss);
  k_final<<<1, 256, 0, stream>>>(rowloss, out);
}
